// FTCA_18820546691547
// MI455X (gfx1250) — compile-verified, matrix operand order checked by reading, not run on hardware
//
#include <hip/hip_runtime.h>
#include <hip/hip_bf16.h>
#include <math.h>
#include <stdint.h>

// ---------------- problem constants ----------------
#define BSZ   16
#define CCH   192
#define HH    120
#define WW    120
#define HW    14400           // 120*120
#define CHW   2764800         // 192*14400
#define AN    36
#define MID   18
#define YCH   72              // 2*AN
#define PB    6912            // pool buffer per batch: 192*36
#define FPB   518400          // AN*HW per batch
#define EPS   1e-5f

typedef __attribute__((ext_vector_type(16))) __bf16        v16bf;
typedef __attribute__((ext_vector_type(8)))  float         v8f;
typedef __attribute__((ext_vector_type(4)))  unsigned int  v4u;
typedef __attribute__((ext_vector_type(8)))  unsigned int  v8u;

// ---------------------------------------------------------------------------
// Kernel 1: 6x6 exact average pool of x  -> poolbuf (bs, 192*36) flat.
// 'agent' (bs,36,192) is just poolbuf reinterpreted row-major (torch .view).
// ---------------------------------------------------------------------------
__global__ void k_pool(const float* __restrict__ x, float* __restrict__ poolbuf) {
    __shared__ float cell[AN];
    const int plane = blockIdx.x;            // b*192 + c
    const int b = plane / CCH, c = plane % CCH;
    const float* p = x + (size_t)b * CHW + (size_t)c * HW;
    if (threadIdx.x < AN) cell[threadIdx.x] = 0.0f;
    __syncthreads();
    for (int n = threadIdx.x; n < HW; n += blockDim.x) {
        const int py = n / WW, px = n % WW;
        const int a = (py / 20) * 6 + (px / 20);
        atomicAdd(&cell[a], p[n]);           // ds_add_f32
    }
    __syncthreads();
    if (threadIdx.x < AN)
        poolbuf[(size_t)b * PB + c * AN + threadIdx.x] = cell[threadIdx.x] * (1.0f / 400.0f);
}

// ---------------------------------------------------------------------------
// Kernel 2: v = conv1x1(pool6(x), val_w)   (pool and 1x1 conv commute)
// Pool buffer staged into LDS via ASYNC global->LDS copies (ASYNCcnt path).
// ---------------------------------------------------------------------------
__global__ void k_val(const float* __restrict__ poolbuf,
                      const float* __restrict__ val_w,
                      float* __restrict__ v) {
    __shared__ float ps[PB];
    const int b = blockIdx.x;

    // async-fill LDS with poolbuf[b] : 1728 x b128 transfers issued by the block
    {
        const unsigned long long gbase =
            (unsigned long long)(uintptr_t)(poolbuf + (size_t)b * PB);
        const unsigned lds_base = (unsigned)(uintptr_t)&ps[0];   // low 32b = LDS offset
        for (int i = threadIdx.x; i < PB / 4; i += blockDim.x) {
            const unsigned ldsa = lds_base + (unsigned)i * 16u;
            const unsigned voff = (unsigned)i * 16u;
            asm volatile("global_load_async_to_lds_b128 %0, %1, %2"
                         :: "v"(ldsa), "v"(voff), "s"(gbase) : "memory");
        }
        asm volatile("s_wait_asynccnt 0x0" ::: "memory");
    }
    __syncthreads();

    for (int o = threadIdx.x; o < PB; o += blockDim.x) {
        const int c = o / AN, a = o % AN;
        const float* wrow = val_w + (size_t)c * CCH;
        float acc = 0.0f;
#pragma unroll 8
        for (int ci = 0; ci < CCH; ++ci) acc += wrow[ci] * ps[ci * AN + a];
        v[(size_t)b * PB + o] = acc;
    }
}

// ---------------------------------------------------------------------------
// Kernel 3: grouped 3x3 conv (groups=3) + BN + ReLU -> k1  (fp32)
// ---------------------------------------------------------------------------
__global__ void k_keyconv(const float* __restrict__ x, const float* __restrict__ key_w,
                          const float* __restrict__ g, const float* __restrict__ bta,
                          const float* __restrict__ m, const float* __restrict__ var,
                          float* __restrict__ k1) {
    __shared__ float wl[64 * 9];
    const int plane = blockIdx.y;            // b*192 + co
    const int b = plane / CCH, co = plane % CCH;
    const int grp = co / 64;
    for (int i = threadIdx.x; i < 576; i += blockDim.x) wl[i] = key_w[(size_t)co * 576 + i];
    __syncthreads();

    const float inv  = g[co] * rsqrtf(var[co] + EPS);
    const float beta = bta[co] - m[co] * inv;

    const int n = blockIdx.x * blockDim.x + threadIdx.x;
    if (n >= HW) return;
    const int y0 = n / WW, x0 = n % WW;
    const float* xb = x + (size_t)b * CHW + (size_t)(grp * 64) * HW;

    float acc = 0.0f;
    for (int ci = 0; ci < 64; ++ci) {
        const float* pl = xb + (size_t)ci * HW;
        if (ci + 1 < 64) __builtin_prefetch(pl + HW + n, 0, 1);   // global_prefetch_b8
        const float* wr = wl + ci * 9;
#pragma unroll
        for (int ky = 0; ky < 3; ++ky) {
            const int yy = y0 + ky - 1;
            if ((unsigned)yy >= HH) continue;
#pragma unroll
            for (int kx = 0; kx < 3; ++kx) {
                const int xx = x0 + kx - 1;
                if ((unsigned)xx >= WW) continue;
                acc += pl[yy * WW + xx] * wr[ky * 3 + kx];
            }
        }
    }
    k1[(size_t)b * CHW + (size_t)co * HW + n] = fmaxf(acc * inv + beta, 0.0f);
}

// ----- helper: load 16 contiguous fp32 and convert to a v16bf WMMA fragment
__device__ inline v16bf ld16_bf(const float* p) {
    v16bf r;
#pragma unroll
    for (int i = 0; i < 16; ++i) r[i] = (__bf16)p[i];
    return r;
}

// ---------------------------------------------------------------------------
// Kernel 4: agent_attn scores = agent(36x192) . k1flat(14400x192)^T per batch.
// bf16 WMMA 16x16x32, K=192 in 6 steps. Padded A-rows are CLAMPED (not
// branched): garbage lands only in D rows >=36, which are never stored, so
// the K-loop stays EXEC-uniform with no save/restore.
// ---------------------------------------------------------------------------
__global__ void k_agent_scores(const float* __restrict__ poolbuf,
                               const float* __restrict__ k1,
                               float* __restrict__ y) {
    const int wave = threadIdx.x >> 5;
    const int lane = threadIdx.x & 31;
    const int b = blockIdx.z;
    const int mtile = blockIdx.y;                       // 0..2  (rows 0..47, padded)
    const int ntile = blockIdx.x * 8 + wave;            // 0..899
    if (ntile >= 900) return;
    const int mbase = mtile * 16, nbase = ntile * 16;
    const int koff_lane = ((lane >> 4) << 4);           // 0 or 16
    const int arow = min(mbase + (lane & 15), AN - 1);  // clamp padded rows
    const int bcol = nbase + (lane & 15);
    const float* apb = poolbuf + (size_t)b * PB + (size_t)arow * CCH;
    const float* bpb = k1 + (size_t)b * CHW + (size_t)bcol * CCH;

    v8f acc = {};
#pragma unroll
    for (int s = 0; s < 6; ++s) {
        const int ko = s * 32 + koff_lane;
        v16bf a  = ld16_bf(apb + ko);
        v16bf bb = ld16_bf(bpb + ko);
        acc = __builtin_amdgcn_wmma_f32_16x16x32_bf16(false, a, false, bb,
                                                      (short)0, acc, false, false);
    }
    // D layout: lanes 0-15 -> M=r, lanes 16-31 -> M=r+8 ; N = lane&15
    const int mhi = (lane >> 4) * 8;
#pragma unroll
    for (int r = 0; r < 8; ++r) {
        const int a_idx = mbase + r + mhi;
        if (a_idx < AN)
            y[(size_t)b * (YCH * (size_t)HW) + (size_t)a_idx * HW + nbase + (lane & 15)] = acc[r];
    }
}

// ---------------------------------------------------------------------------
// Kernel 5: in-place softmax along n (14400) for y channels [0..35].
// ---------------------------------------------------------------------------
__global__ void k_rowsoftmax(float* __restrict__ y) {
    __shared__ float red[256];
    const int a = blockIdx.x, b = blockIdx.y;
    float* row = y + (size_t)b * (YCH * (size_t)HW) + (size_t)a * HW;
    const int t = threadIdx.x;

    float mx = -INFINITY;
    for (int n = t; n < HW; n += 256) mx = fmaxf(mx, row[n]);
    red[t] = mx; __syncthreads();
    for (int s = 128; s > 0; s >>= 1) { if (t < s) red[t] = fmaxf(red[t], red[t + s]); __syncthreads(); }
    mx = red[0]; __syncthreads();

    float sum = 0.0f;
    for (int n = t; n < HW; n += 256) sum += __expf(row[n] - mx);
    red[t] = sum; __syncthreads();
    for (int s = 128; s > 0; s >>= 1) { if (t < s) red[t] += red[t + s]; __syncthreads(); }
    const float inv = 1.0f / red[0];

    for (int n = t; n < HW; n += 256) row[n] = __expf(row[n] - mx) * inv;
}

// ---------------------------------------------------------------------------
// Kernel 6: q_attn = softmax_a( xflat(14400x192) . agent(36x192)^T ), written
// transposed into y channels [36..71]. Padded agent columns are CLAMPED; the
// fused softmax overwrites them with -inf so they contribute nothing.
// ---------------------------------------------------------------------------
__global__ void k_qattn(const float* __restrict__ x, const float* __restrict__ poolbuf,
                        float* __restrict__ y) {
    const int wave = threadIdx.x >> 5;
    const int lane = threadIdx.x & 31;
    const int b = blockIdx.y;
    const int mtile = blockIdx.x * 8 + wave;            // pixel tile 0..899
    if (mtile >= 900) return;
    const int mbase = mtile * 16;
    const int koff_lane = ((lane >> 4) << 4);
    const int prow = mbase + (lane & 15);
    const float* apb = x + (size_t)b * CHW + (size_t)prow * CCH;      // q1 raw view row

    const float* bcol_p[3];
#pragma unroll
    for (int tcol = 0; tcol < 3; ++tcol) {
        const int acol = min(tcol * 16 + (lane & 15), AN - 1);        // clamp pads
        bcol_p[tcol] = poolbuf + (size_t)b * PB + (size_t)acol * CCH;
    }

    v8f acc[3] = {{}, {}, {}};
#pragma unroll
    for (int s = 0; s < 6; ++s) {
        const int ko = s * 32 + koff_lane;
        v16bf a = ld16_bf(apb + ko);
#pragma unroll
        for (int tcol = 0; tcol < 3; ++tcol) {
            v16bf bb = ld16_bf(bcol_p[tcol] + ko);
            acc[tcol] = __builtin_amdgcn_wmma_f32_16x16x32_bf16(false, a, false, bb,
                                                                (short)0, acc[tcol], false, false);
        }
    }

    const int mhi = (lane >> 4) * 8;
    const int colpad = (lane & 15);
#pragma unroll
    for (int r = 0; r < 8; ++r) {
        float v0 = acc[0][r];
        float v1 = acc[1][r];
        float v2 = (colpad < AN - 32) ? acc[2][r] : -INFINITY;   // pad cols 36..47
        float mx = fmaxf(fmaxf(v0, v1), v2);
#pragma unroll
        for (int off = 8; off >= 1; off >>= 1) mx = fmaxf(mx, __shfl_xor(mx, off, 32));
        float e0 = __expf(v0 - mx), e1 = __expf(v1 - mx);
        float e2 = (colpad < AN - 32) ? __expf(v2 - mx) : 0.0f;
        float sm = e0 + e1 + e2;
#pragma unroll
        for (int off = 8; off >= 1; off >>= 1) sm += __shfl_xor(sm, off, 32);
        const float inv = 1.0f / sm;
        const int n = mbase + r + mhi;                  // pixel index
        float* yb = y + (size_t)b * (YCH * (size_t)HW);
        yb[(size_t)(AN + 0 * 16 + colpad) * HW + n] = e0 * inv;
        yb[(size_t)(AN + 1 * 16 + colpad) * HW + n] = e1 * inv;
        if (colpad < AN - 32) yb[(size_t)(AN + 2 * 16 + colpad) * HW + n] = e2 * inv;
    }
}

// ---------------------------------------------------------------------------
// Kernel 7: attention-map chain, fused per pixel; mean over the 9 kernel
// positions pre-folded into an effective 36x18 weight. Writes F (a-major).
// ---------------------------------------------------------------------------
__global__ void k_attchain(const float* __restrict__ y,
                           const float* __restrict__ w1,
                           const float* __restrict__ g2, const float* __restrict__ b2,
                           const float* __restrict__ m2, const float* __restrict__ v2,
                           const float* __restrict__ w2, const float* __restrict__ bias2,
                           float* __restrict__ F) {
    __shared__ float w1s[MID * YCH];     // 1296
    __shared__ float weff[AN * MID];     // 648  (pre-averaged att_w2)
    __shared__ float beff[AN];
    __shared__ float inv2[MID], bet2[MID];
    const int b = blockIdx.y;
    for (int i = threadIdx.x; i < MID * YCH; i += blockDim.x) w1s[i] = w1[i];
    for (int i = threadIdx.x; i < AN * MID; i += blockDim.x) {
        const int a = i / MID, mm = i % MID;
        float s = 0.0f;
#pragma unroll
        for (int j = 0; j < 9; ++j) s += w2[(size_t)(a * 9 + j) * MID + mm];
        weff[i] = s * (1.0f / 9.0f);
    }
    if (threadIdx.x < AN) {
        float s = 0.0f;
#pragma unroll
        for (int j = 0; j < 9; ++j) s += bias2[threadIdx.x * 9 + j];
        beff[threadIdx.x] = s * (1.0f / 9.0f);
    }
    if (threadIdx.x < MID) {
        const int mm = threadIdx.x;
        const float iv = g2[mm] * rsqrtf(v2[mm] + EPS);
        inv2[mm] = iv;  bet2[mm] = b2[mm] - m2[mm] * iv;
    }
    __syncthreads();

    const int n = blockIdx.x * blockDim.x + threadIdx.x;
    if (n >= HW) return;
    const float* yb = y + (size_t)b * (YCH * (size_t)HW) + n;

    float mid[MID];
#pragma unroll
    for (int mm = 0; mm < MID; ++mm) mid[mm] = 0.0f;
    for (int ch = 0; ch < YCH; ++ch) {
        const float yv = yb[(size_t)ch * HW];
#pragma unroll
        for (int mm = 0; mm < MID; ++mm) mid[mm] += w1s[mm * YCH + ch] * yv;
    }
#pragma unroll
    for (int mm = 0; mm < MID; ++mm) mid[mm] = fmaxf(mid[mm] * inv2[mm] + bet2[mm], 0.0f);

    float* Fb = F + (size_t)b * FPB + n;
    for (int a = 0; a < AN; ++a) {
        float acc = beff[a];
#pragma unroll
        for (int mm = 0; mm < MID; ++mm) acc += weff[a * MID + mm] * mid[mm];
        Fb[(size_t)a * HW] = acc;
    }
}

// ---------------------------------------------------------------------------
// Kernel 8: final stage. v (192x36 contiguous, 27KB) staged into LDS with a
// single Tensor-Data-Mover descriptor: wave0 builds D# groups 0/1 in SGPRs,
// issues one tensor_load_to_lds, waits on TENSORcnt, block syncs.
// Then: softmax over contiguous 36-groups of F; k2 . v^T ; + k1 residual.
// ---------------------------------------------------------------------------
__global__ void k_final(const float* __restrict__ F, const float* __restrict__ v,
                        const float* __restrict__ k1, float* __restrict__ out) {
    __shared__ float vsm[PB];            // 192x36, 27KB
    const int b = blockIdx.y;

    if (threadIdx.x < 32) {              // one wave drives the TDM copy
        const unsigned long long ga =
            (unsigned long long)(uintptr_t)(v + (size_t)b * PB);
        const unsigned lds = (unsigned)(uintptr_t)&vsm[0];       // low 32b = LDS offset

        v4u g0;
        g0[0] = 1u;                                              // count=1, user mode
        g0[1] = lds;                                             // lds_addr
        g0[2] = (unsigned)(ga & 0xFFFFFFFFull);                  // global_addr[31:0]
        g0[3] = (unsigned)((ga >> 32) & 0x01FFFFFFull)           // global_addr[56:32]
              | 0x80000000u;                                     // type=2 ("image")

        v8u g1;
        g1[0] = (2u << 16);                                      // data_size=4B, no mask
        g1[1] = ((unsigned)PB & 0xFFFFu) << 16;                  // tensor_dim0 lo16
        g1[2] = ((unsigned)PB >> 16) | (1u << 16);               // dim0 hi16 | dim1 lo16=1
        g1[3] = ((unsigned)PB & 0xFFFFu) << 16;                  // dim1 hi16=0 | tile_dim0
        g1[4] = 1u;                                              // tile_dim1=1, tile_dim2=0
        g1[5] = (unsigned)PB;                                    // dim0_stride lo32
        g1[6] = 0u;                                              // stride0 hi | stride1 lo
        g1[7] = 0u;                                              // stride1 hi
        asm volatile("tensor_load_to_lds %0, %1"
                     :: "s"(g0), "s"(g1) : "memory");
        __builtin_amdgcn_s_wait_tensorcnt(0);
    }
    __syncthreads();

    const int n2 = blockIdx.x * blockDim.x + threadIdx.x;
    if (n2 >= HW) return;
    const float* fp = F + (size_t)b * FPB + (size_t)n2 * AN;   // 36 contiguous

    float e[AN];
    float mx = -INFINITY;
#pragma unroll
    for (int a = 0; a < AN; ++a) { e[a] = fp[a]; mx = fmaxf(mx, e[a]); }
    float sm = 0.0f;
#pragma unroll
    for (int a = 0; a < AN; ++a) { e[a] = __expf(e[a] - mx); sm += e[a]; }
    const float inv = 1.0f / sm;

    const float* k1b = k1 + (size_t)b * CHW + n2;
    float* ob = out + (size_t)b * CHW + n2;
    for (int c = 0; c < CCH; ++c) {
        const float* vr = vsm + c * AN;
        float acc = 0.0f;
#pragma unroll
        for (int a = 0; a < AN; ++a) acc += e[a] * vr[a];
        ob[(size_t)c * HW] = acc * inv + k1b[(size_t)c * HW];
    }
}

// ---------------------------------------------------------------------------
extern "C" void kernel_launch(void* const* d_in, const int* in_sizes, int n_in,
                              void* d_out, int out_size, void* d_ws, size_t ws_size,
                              hipStream_t stream) {
    const float* x      = (const float*)d_in[0];
    const float* key_w  = (const float*)d_in[1];
    const float* bn1_g  = (const float*)d_in[2];
    const float* bn1_b  = (const float*)d_in[3];
    const float* bn1_m  = (const float*)d_in[4];
    const float* bn1_v  = (const float*)d_in[5];
    const float* val_w  = (const float*)d_in[6];
    const float* att_w1 = (const float*)d_in[7];
    const float* bn2_g  = (const float*)d_in[8];
    const float* bn2_b  = (const float*)d_in[9];
    const float* bn2_m  = (const float*)d_in[10];
    const float* bn2_v  = (const float*)d_in[11];
    const float* att_w2 = (const float*)d_in[12];
    const float* att_b2 = (const float*)d_in[13];
    float* out = (float*)d_out;

    // workspace carve-up (floats)
    float* ws   = (float*)d_ws;
    float* k1   = ws;                                   // 44,236,800
    float* y    = k1 + (size_t)BSZ * CHW;               // 16,588,800
    float* F    = y  + (size_t)BSZ * YCH * HW;          //  8,294,400
    float* pool = F  + (size_t)BSZ * FPB;               //    110,592
    float* v    = pool + (size_t)BSZ * PB;              //    110,592

    k_pool<<<dim3(BSZ * CCH), 256, 0, stream>>>(x, pool);
    k_val<<<dim3(BSZ), 256, 0, stream>>>(pool, val_w, v);
    k_keyconv<<<dim3((HW + 255) / 256, BSZ * CCH), 256, 0, stream>>>(
        x, key_w, bn1_g, bn1_b, bn1_m, bn1_v, k1);
    k_agent_scores<<<dim3(113, 3, BSZ), 256, 0, stream>>>(pool, k1, y);
    k_rowsoftmax<<<dim3(AN, BSZ), 256, 0, stream>>>(y);
    k_qattn<<<dim3(113, BSZ), 256, 0, stream>>>(x, pool, y);
    k_attchain<<<dim3((HW + 255) / 256, BSZ), 256, 0, stream>>>(
        y, att_w1, bn2_g, bn2_b, bn2_m, bn2_v, att_w2, att_b2, F);
    k_final<<<dim3((HW + 255) / 256, BSZ), 256, 0, stream>>>(F, v, k1, out);
}
